// MultiHeadAttention_13340168421416
// MI455X (gfx1250) — matrix operand order repaired and checked by reading, not compiled, not run on hardware
//
#include <hip/hip_runtime.h>
#include <math.h>

typedef _Float16 h16;
typedef __attribute__((ext_vector_type(8)))  _Float16 v8h;
typedef __attribute__((ext_vector_type(16))) _Float16 v16h;
typedef __attribute__((ext_vector_type(8)))  float    v8f;

// ---------------------------------------------------------------------------
// Tensor Data Mover (TDM) async 2-D tile load: global -> LDS.
// Descriptor built per CDNA5 ISA.
// This toolchain exposes the 6-arg builtin:
//   (uint32x4 g0, int32x8 g1, int32x4 g2, int32x4 g3, int32x8 aux, i32 cpol)
// ---------------------------------------------------------------------------
#if defined(__has_builtin)
#if __has_builtin(__builtin_amdgcn_tensor_load_to_lds) && \
    __has_builtin(__builtin_amdgcn_s_wait_tensorcnt)
#define USE_TDM 1
#endif
#endif

#ifdef USE_TDM
typedef unsigned int u32x4 __attribute__((ext_vector_type(4)));
typedef int          i32x8 __attribute__((ext_vector_type(8)));
typedef int          i32x4 __attribute__((ext_vector_type(4)));

// Load a [tile_h x tile_w] f16 tile (row stride = row_stride elements) into
// LDS at ldst (rows packed contiguously). Issued once per calling wave;
// completion tracked by TENSORcnt.
__device__ __forceinline__ void tdm_load_2d_f16(const h16* gsrc, h16* ldst,
                                                int tile_w, int tile_h,
                                                long row_stride) {
  unsigned long long ga = (unsigned long long)(uintptr_t)gsrc;
  unsigned la = (unsigned)(uintptr_t)ldst;  // flat addr[31:0] == LDS byte addr
  u32x4 g0;
  g0[0] = 1u;                                   // count=1, user descriptor
  g0[1] = la;                                   // lds_addr
  g0[2] = (unsigned)(ga & 0xFFFFFFFFu);         // global_addr[31:0]
  g0[3] = (unsigned)((ga >> 32) & 0x1FFFFFFu)   // global_addr[56:32]
          | (2u << 30);                         // type = 2 ("image")
  i32x8 g1;
  g1[0] = 1 << 16;                              // data_size = 1 -> 2 bytes
  g1[1] = (int)((unsigned)(tile_w & 0xFFFF) << 16);  // tensor_dim0[15:0]
  g1[2] = (int)((unsigned)(tile_h & 0xFFFF) << 16);  // dim0 hi=0 | tensor_dim1 lo
  g1[3] = (int)((unsigned)(tile_w & 0xFFFF) << 16);  // dim1 hi=0 | tile_dim0
  g1[4] = tile_h & 0xFFFF;                      // tile_dim1 | tile_dim2=0
  g1[5] = (int)(unsigned)(row_stride & 0xFFFFFFFFu); // tensor_dim0_stride lo32
  g1[6] = (int)((row_stride >> 32) & 0xFFFF);   // stride0 hi16 | stride1 lo=0
  g1[7] = 0;
  i32x4 z4 = {0, 0, 0, 0};                      // groups 2/3: 2-D tensor
  i32x8 z8 = {0, 0, 0, 0, 0, 0, 0, 0};          // aux group (zero-filled)
  __builtin_amdgcn_tensor_load_to_lds(g0, g1, z4, z4, z8, 0);
}
#endif

// ---------------------------------------------------------------------------
// WMMA helpers (CDNA5 wave32, V_WMMA_F32_16X16X32_F16)
// ---------------------------------------------------------------------------
__device__ __forceinline__ v8f wmma16(v16h a, v16h b, v8f c) {
  return __builtin_amdgcn_wmma_f32_16x16x32_f16(false, a, false, b, (short)0, c,
                                                false, false);
}

// A-matrix fragment (16x32 f16, row-major source, lda in elements).
__device__ __forceinline__ v16h load_a_frag(const h16* __restrict__ base,
                                            int lda, int k0) {
  int lane = threadIdx.x & 31;
  int row  = lane & 15;
  int g    = lane >> 4;
  const h16* p = base + (long)row * lda + k0 + g * 16;
  return *(const v16h*)p;
}

// B-matrix fragment (32x16 f16) from a K-major matrix Bt[N][K].
__device__ __forceinline__ v16h load_b_frag(const h16* __restrict__ bt,
                                            long ldb, int n0, int k0) {
  int lane = threadIdx.x & 31;
  int n    = n0 + (lane & 15);
  int g    = lane >> 4;
  return *(const v16h*)(bt + (long)n * ldb + k0 + g * 16);
}

// ---------------------------------------------------------------------------
// f32 -> f16 convert / transpose-convert staging kernels
// ---------------------------------------------------------------------------
__global__ __launch_bounds__(256) void convert_f16_kernel(
    const float* __restrict__ src, h16* __restrict__ dst, int n) {
  int i = blockIdx.x * 256 + threadIdx.x;
  if (i < n) dst[i] = (h16)src[i];
}

// src: [batch][rows][cols] f32  ->  dst: [batch][cols][rows] f16
__global__ __launch_bounds__(256) void transpose_convert_kernel(
    const float* __restrict__ src, h16* __restrict__ dst,
    int rows, int cols, int total) {
  int i = blockIdx.x * 256 + threadIdx.x;
  if (i >= total) return;
  int rc = rows * cols;
  int bi = i / rc;
  int r  = (i % rc) / cols;
  int c  = i % cols;
  dst[(long)bi * rc + (long)c * rows + r] = (h16)src[i];
}

// ---------------------------------------------------------------------------
// Batched WMMA GEMM with double-buffered LDS staging of the 128x32 B slab.
// With TDM: wave 0 DMAs each slab into LDS (tensor_load_to_lds) while all
// waves run WMMAs on the previous slab; completion via s_wait_tensorcnt +
// barrier. Fallback: cooperative global->reg->ds_store pipeline.
// ---------------------------------------------------------------------------
template <bool OUT_F16, bool TRANS_OUT>
__global__ __launch_bounds__(256) void gemm_wmma_kernel(
    const h16* __restrict__ A, const h16* __restrict__ Bt,
    const float* __restrict__ bias, void* __restrict__ Cout,
    int M, int N, int K, int nH,
    long aStride, long bStride, long cStride, int biasStride) {
  __shared__ __align__(32) h16 s_b[2][128 * 32];  // 16KB double buffer

  int tilesM = M >> 6;
  int tilesN = N >> 7;
  int tilesPerBatch = tilesM * tilesN;
  int batch = blockIdx.x / tilesPerBatch;
  int t     = blockIdx.x % tilesPerBatch;
  int row0  = (t % tilesM) << 6;
  int col0  = (t / tilesM) << 7;

  const h16*   Ab = A  + (long)(batch / nH) * aStride;
  const h16*   Bb = Bt + (long)(batch % nH) * bStride;
  const float* bb = bias + (long)(batch % nH) * biasStride;

  int wave = threadIdx.x >> 5;
  int lane = threadIdx.x & 31;
  int r0   = row0 + ((wave & 3) << 4);
  int c0l  = (wave >> 2) << 6;  // column base within the 128 tile

  v8f acc[4];
#pragma unroll
  for (int s = 0; s < 4; ++s)
#pragma unroll
    for (int v = 0; v < 8; ++v) acc[s][v] = 0.0f;

  int nsteps = K >> 5;  // >= 2 here (K = 512 or 4096)
  const h16* Arow = Ab + (long)r0 * K;
  const h16* Btile = Bb + (long)col0 * K;  // slab origin: row stride K

#ifdef USE_TDM
  if (wave == 0)  // prologue: DMA slab 0
    tdm_load_2d_f16(Btile, &s_b[0][0], 32, 128, K);
  for (int s = 0; s < nsteps; ++s) {
    if (wave == 0) {
      if (s + 1 < nsteps) {
        // prior barrier guarantees everyone finished reading buf[(s+1)&1]
        tdm_load_2d_f16(Btile + ((s + 1) << 5), &s_b[(s + 1) & 1][0],
                        32, 128, K);
        __builtin_amdgcn_s_wait_tensorcnt(1);  // slab s complete (in-order)
      } else {
        __builtin_amdgcn_s_wait_tensorcnt(0);
      }
    }
    __syncthreads();  // publish slab s
    v16h a = load_a_frag(Arow, K, s << 5);
    const h16* bbuf = &s_b[s & 1][0];
    int koff = (lane >> 4) << 4;
#pragma unroll
    for (int q = 0; q < 4; ++q) {
      int nl = c0l + q * 16 + (lane & 15);
      v16h b = *(const v16h*)(bbuf + nl * 32 + koff);  // ds_load_b128 x2
      acc[q] = wmma16(a, b, acc[q]);
    }
    __syncthreads();  // all reads done before next DMA overwrites
  }
#else
  // cooperative loader: thread covers one v16h (32B); 256 x 32B = 8KB slab
  int ln_row = threadIdx.x >> 1;
  int ln_k   = (threadIdx.x & 1) << 4;
  const h16* bsrc = Btile + (long)ln_row * K + ln_k;
  h16* bdst0 = &s_b[0][ln_row * 32 + ln_k];
  h16* bdst1 = &s_b[1][ln_row * 32 + ln_k];
  v16h stage = *(const v16h*)(bsrc);
  *(v16h*)bdst0 = stage;
  stage = *(const v16h*)(bsrc + 32);
  for (int s = 0; s < nsteps; ++s) {
    __syncthreads();
    if (s + 1 < nsteps) {
      *(v16h*)((s & 1) ? bdst0 : bdst1) = stage;
      if (s + 2 < nsteps) stage = *(const v16h*)(bsrc + ((s + 2) << 5));
    }
    v16h a = load_a_frag(Arow, K, s << 5);
    const h16* bbuf = &s_b[s & 1][0];
    int koff = (lane >> 4) << 4;
#pragma unroll
    for (int q = 0; q < 4; ++q) {
      int nl = c0l + q * 16 + (lane & 15);
      v16h b = *(const v16h*)(bbuf + nl * 32 + koff);
      acc[q] = wmma16(a, b, acc[q]);
    }
  }
#endif

  // C/D layout: lane L -> col L%16, rows (L/16)*8 + v
  int col = lane & 15;
  int rb  = (lane >> 4) << 3;
#pragma unroll
  for (int s = 0; s < 4; ++s) {
    int n = col0 + c0l + s * 16 + col;
    float bv = bb[n];
#pragma unroll
    for (int v = 0; v < 8; ++v) {
      int m = r0 + rb + v;
      float val = acc[s][v] + bv;
      if constexpr (OUT_F16) {
        h16* C = (h16*)Cout + (long)batch * cStride;
        if constexpr (TRANS_OUT)
          C[(long)n * M + m] = (h16)val;  // store V transposed [N][M]
        else
          C[(long)m * N + n] = (h16)val;
      } else {
        float* C = (float*)Cout + (long)batch * cStride;
        C[(long)m * N + n] = val;
      }
    }
  }
}

// ---------------------------------------------------------------------------
// Flash-style attention. One WG (8 wave32) per (b, h, 16 query rows).
// Q,K: [B,H,S,E] f16.  Vt: [B,H,E,S] f16.  heads: [B,S,H*E] f16.
// ---------------------------------------------------------------------------
__global__ __launch_bounds__(256) void attention_kernel(
    const h16* __restrict__ Q, const h16* __restrict__ Kt,
    const h16* __restrict__ Vt, h16* __restrict__ heads, float scale) {
  constexpr int S = 1024, E = 512, HE = 4096;

  __shared__ __align__(32) h16   s_q[16][512];       // 16KB Q tile
  __shared__ __align__(32) float s_scores[16][128];  // 8KB
  __shared__ __align__(32) h16   s_p[16][128];       // 4KB
  __shared__ float s_red[16][8];
  __shared__ float s_m[16], s_l[16], s_alpha[16];

  int qt = blockIdx.x & 63;  // S/16 tiles
  int bh = blockIdx.x >> 6;  // b*H + h
  int b  = bh >> 3;
  int h  = bh & 7;
  int q0 = qt << 4;

  const h16* Qb = Q  + (long)bh * S * E + (long)q0 * E;
  const h16* Kb = Kt + (long)bh * S * E;
  const h16* Vb = Vt + (long)bh * E * S;

  int tid  = threadIdx.x;
  int wave = tid >> 5;
  int lane = tid & 31;
  int col  = lane & 15;
  int rb   = (lane >> 4) << 3;

  // stage the 16x512 Q tile into LDS
#ifdef USE_TDM
  if (wave == 0) {
    tdm_load_2d_f16(Qb, &s_q[0][0], 512, 16, 512);
    __builtin_amdgcn_s_wait_tensorcnt(0);
  }
#else
  {
    int flat = tid << 5;  // 32 halfs per thread
    const v16h* src = (const v16h*)(Qb + flat);
    v16h a0 = src[0];
    v16h a1 = src[1];
    v16h* dst = (v16h*)(&s_q[0][0] + flat);
    dst[0] = a0;
    dst[1] = a1;
  }
#endif
  if (tid < 16) { s_m[tid] = -3.0e38f; s_l[tid] = 0.0f; }
  __syncthreads();

  v8f acc[4];
#pragma unroll
  for (int s = 0; s < 4; ++s)
#pragma unroll
    for (int v = 0; v < 8; ++v) acc[s][v] = 0.0f;

  int sr = tid >> 3;         // softmax row (only <16 active)
  int sc8 = (tid & 7) << 4;  // softmax 16-col slab base

  for (int kb = 0; kb < S; kb += 128) {
    // ---- stage 1: score tile [16q x 16k], keys kb + wave*16 ----
    v8f sc;
#pragma unroll
    for (int v = 0; v < 8; ++v) sc[v] = 0.0f;
    int keyBase = kb + (wave << 4);
    if (kb + 128 < S)  // warm L2/WGP$ one chunk ahead (global_prefetch_b8)
      __builtin_prefetch(Kb + (long)(kb + 128 + (wave << 4) + col) * E, 0, 1);
#pragma unroll 4
    for (int es = 0; es < E; es += 32) {
      const h16* qrow = &s_q[lane & 15][0];
      int g16 = (lane >> 4) << 4;
      v16h a = *(const v16h*)(qrow + es + g16);
      v16h bf = load_b_frag(Kb, E, keyBase, es);
      sc = wmma16(a, bf, sc);
    }
#pragma unroll
    for (int v = 0; v < 8; ++v)
      s_scores[rb + v][(wave << 4) + col] = sc[v] * scale;
    __syncthreads();

    // ---- stage 2: parallel online softmax (128 threads, LDS reductions) ----
    if (tid < 128) {
      float lm = -3.0e38f;
#pragma unroll
      for (int j = 0; j < 16; ++j) lm = fmaxf(lm, s_scores[sr][sc8 + j]);
      s_red[sr][tid & 7] = lm;
    }
    __syncthreads();
    if (tid < 16) {
      float mnew = s_m[tid];
#pragma unroll
      for (int j = 0; j < 8; ++j) mnew = fmaxf(mnew, s_red[tid][j]);
      s_alpha[tid] = __expf(s_m[tid] - mnew);
      s_m[tid] = mnew;
    }
    __syncthreads();
    if (tid < 128) {
      float mnew = s_m[sr];
      float lsum = 0.0f;
#pragma unroll
      for (int j = 0; j < 16; ++j) {
        float p = __expf(s_scores[sr][sc8 + j] - mnew);
        s_p[sr][sc8 + j] = (h16)p;
        lsum += p;
      }
      s_red[sr][tid & 7] = lsum;
    }
    __syncthreads();
    if (tid < 16) {
      float l = s_l[tid] * s_alpha[tid];
#pragma unroll
      for (int j = 0; j < 8; ++j) l += s_red[tid][j];
      s_l[tid] = l;
    }

    // ---- stage 3: rescale accumulators, then P @ V for cols wave*64.. ----
#pragma unroll
    for (int s = 0; s < 4; ++s)
#pragma unroll
      for (int v = 0; v < 8; ++v) acc[s][v] *= s_alpha[rb + v];

    int c0 = wave << 6;
    for (int ks = 0; ks < 128; ks += 32) {
      const h16* prow = &s_p[lane & 15][0];
      int g16 = (lane >> 4) << 4;
      v16h pa = *(const v16h*)(prow + ks + g16);
#pragma unroll
      for (int s = 0; s < 4; ++s) {
        v16h vf = load_b_frag(Vb, S, c0 + s * 16, kb + ks);
        acc[s] = wmma16(pa, vf, acc[s]);
      }
    }
    __syncthreads();
  }

  // ---- epilogue: divide by l, write heads[b, q, h*E + e] as f16 ----
  float inv[8];
#pragma unroll
  for (int v = 0; v < 8; ++v) inv[v] = 1.0f / s_l[rb + v];
  int c0 = wave << 6;
  h16* Hb = heads + ((long)b * S + q0) * HE + h * E;
#pragma unroll
  for (int s = 0; s < 4; ++s)
#pragma unroll
    for (int v = 0; v < 8; ++v)
      Hb[(long)(rb + v) * HE + c0 + s * 16 + col] = (h16)(acc[s][v] * inv[v]);
}

// ---------------------------------------------------------------------------
// Host-side orchestration (graph-capture safe: only kernel launches on stream)
// ---------------------------------------------------------------------------
extern "C" void kernel_launch(void* const* d_in, const int* in_sizes, int n_in,
                              void* d_out, int out_size, void* d_ws,
                              size_t ws_size, hipStream_t stream) {
  (void)in_sizes; (void)n_in; (void)out_size; (void)ws_size;
  constexpr int B = 8, S = 1024, D = 512, E = 512, H = 8;

  const float* input1 = (const float*)d_in[0];
  const float* input2 = (const float*)d_in[1];
  const float* input3 = (const float*)d_in[2];
  const float* Wq = (const float*)d_in[3];
  const float* bq = (const float*)d_in[4];
  const float* Wk = (const float*)d_in[5];
  const float* bk = (const float*)d_in[6];
  const float* Wv = (const float*)d_in[7];
  const float* bv = (const float*)d_in[8];
  const float* Wo = (const float*)d_in[9];
  const float* bo = (const float*)d_in[10];

  size_t off = 0;
  auto alloc = [&](size_t bytes) -> void* {
    void* p = (char*)d_ws + off;
    off += (bytes + 255) & ~(size_t)255;
    return p;
  };

  h16* x1  = (h16*)alloc((size_t)B * S * D * 2);
  h16* x2  = (h16*)alloc((size_t)B * S * D * 2);
  h16* x3  = (h16*)alloc((size_t)B * S * D * 2);
  h16* wqT = (h16*)alloc((size_t)H * D * E * 2);  // [H][E][D]
  h16* wkT = (h16*)alloc((size_t)H * D * E * 2);
  h16* wvT = (h16*)alloc((size_t)H * D * E * 2);
  h16* woT = (h16*)alloc((size_t)H * E * E * 2);  // [512][4096]
  h16* q16 = (h16*)alloc((size_t)B * H * S * E * 2);  // [B,H,S,E]
  h16* k16 = (h16*)alloc((size_t)B * H * S * E * 2);  // [B,H,S,E]
  h16* vT  = (h16*)alloc((size_t)B * H * E * S * 2);  // [B,H,E,S]
  h16* hds = (h16*)alloc((size_t)B * S * H * E * 2);  // [B,S,H*E]

  // 1) stage f16 copies
  int nx = B * S * D;
  convert_f16_kernel<<<(nx + 255) / 256, 256, 0, stream>>>(input1, x1, nx);
  convert_f16_kernel<<<(nx + 255) / 256, 256, 0, stream>>>(input2, x2, nx);
  convert_f16_kernel<<<(nx + 255) / 256, 256, 0, stream>>>(input3, x3, nx);
  int nw = H * D * E;
  transpose_convert_kernel<<<(nw + 255) / 256, 256, 0, stream>>>(Wq, wqT, D, E, nw);
  transpose_convert_kernel<<<(nw + 255) / 256, 256, 0, stream>>>(Wk, wkT, D, E, nw);
  transpose_convert_kernel<<<(nw + 255) / 256, 256, 0, stream>>>(Wv, wvT, D, E, nw);
  int nwo = H * E * E;  // 4096 x 512
  transpose_convert_kernel<<<(nwo + 255) / 256, 256, 0, stream>>>(Wo, woT, H * E, E, nwo);

  // 2) QKV projections: per (b,h) GEMM  [S,D] x [D,E] (+bias), f16 out
  {
    int M = S, N = E, K = D;
    int tiles = (M / 64) * (N / 128);  // 64
    int grid = tiles * B * H;          // 4096
    long aS = (long)S * D, bS = (long)E * D, cS = (long)S * E;
    gemm_wmma_kernel<true, false><<<grid, 256, 0, stream>>>(
        x1, wqT, bq, q16, M, N, K, H, aS, bS, cS, E);
    gemm_wmma_kernel<true, false><<<grid, 256, 0, stream>>>(
        x2, wkT, bk, k16, M, N, K, H, aS, bS, cS, E);
    gemm_wmma_kernel<true, true><<<grid, 256, 0, stream>>>(
        x3, wvT, bv, vT, M, N, K, H, aS, bS, cS, E);  // V stored [E][S]
  }

  // 3) fused softmax(QK^T/sqrt(E)) V -> heads
  attention_kernel<<<B * H * (S / 16), 256, 0, stream>>>(
      q16, k16, vT, hds, 1.0f / sqrtf((float)E));

  // 4) output projection: [B*S, 4096] x [4096, 512] + bo -> f32 d_out
  {
    int M = B * S, N = E, K = H * E;  // 8192, 512, 4096
    int grid = (M / 64) * (N / 128);  // 512
    gemm_wmma_kernel<false, false><<<grid, 256, 0, stream>>>(
        hds, woT, bo, d_out, M, N, K, 1, 0, 0, 0, 0);
  }
}
